// Encoder1DBlock_11184094839395
// MI455X (gfx1250) — compile-verified, matrix operand order checked by reading, not run on hardware
//
#include <hip/hip_runtime.h>
#include <hip/hip_bf16.h>
#include <math.h>

typedef __attribute__((ext_vector_type(16))) _Float16 v16h;
typedef __attribute__((ext_vector_type(8)))  float    v8f;

#define B_  16
#define N_  577
#define C_  1024
#define H_  16
#define D_  64
#define M_  24
#define CH_ 256
#define NSP 576            // spatial tokens per image (24x24)
#define BN  (B_*N_)        // 9232 token rows
#define MSP (B_*NSP)       // 9216 conv pixels

// ---------- fragment index helpers (ISA 7.12.2, wave32, 16-bit 16x32 A/B) ----------
// VGPR i of a 16-half fragment holds the K-pair at:
//   k = (i>=4 ? 16:0) + (i&3)*2 + (lane>=16 ? 8:0)
// -> dwords [4h..4h+3] and [8+4h..8+4h+3]: two B128 loads.
__device__ __forceinline__ int koff(int i, int h){ return ((i>>2)<<4) + ((i&3)<<1) + (h<<3); }

union AF { v16h h; unsigned int u[8]; };
union FA { v8f  v; float f[8]; };

__device__ __forceinline__ void load_frag(AF& f, const _Float16* p, int hf){
  const uint4* p4 = (const uint4*)p;         // p = row base + k0 (16B aligned)
  uint4 a = p4[hf];
  uint4 b = p4[2 + hf];
  f.u[0]=a.x; f.u[1]=a.y; f.u[2]=a.z; f.u[3]=a.w;
  f.u[4]=b.x; f.u[5]=b.y; f.u[6]=b.z; f.u[7]=b.w;
}

__device__ __forceinline__ v8f wmma32(v16h a, v16h b, v8f c){
  return __builtin_amdgcn_wmma_f32_16x16x32_f16(false, a, false, b, (short)0, c, false, false);
}

// ---------- wave reductions via ds_swizzle (group-of-32 xor pattern) ----------
template<int MASK>
__device__ __forceinline__ float swz(float v){
  int r = __builtin_amdgcn_ds_swizzle(__builtin_bit_cast(int, v), (MASK << 10) | 0x1F);
  return __builtin_bit_cast(float, r);
}
__device__ __forceinline__ float redmax16(float v){
  v = fmaxf(v, swz<1>(v)); v = fmaxf(v, swz<2>(v));
  v = fmaxf(v, swz<4>(v)); v = fmaxf(v, swz<8>(v));
  return v;
}
__device__ __forceinline__ float redsum16(float v){
  v += swz<1>(v); v += swz<2>(v); v += swz<4>(v); v += swz<8>(v);
  return v;
}
__device__ __forceinline__ float redsum32(float v){
  v += swz<1>(v); v += swz<2>(v); v += swz<4>(v); v += swz<8>(v); v += swz<16>(v);
  return v;
}

__device__ __forceinline__ float gelu_tanh(float x){
  float x3 = x*x*x;
  return 0.5f*x*(1.0f + tanhf(0.7978845608028654f*(x + 0.044715f*x3)));
}

// ============================ weight prep ============================
__global__ __launch_bounds__(256) void cast_f16_kernel(const float* __restrict__ in,
                                                       _Float16* __restrict__ out, int n){
  int i = blockIdx.x*256 + threadIdx.x;
  if (i < n) out[i] = (_Float16)in[i];
}

// conv w: [O][Cin][3][3] f32 -> [O][9*Cin] f16 with k = tap*Cin + c
__global__ __launch_bounds__(256) void prep_convw_kernel(const float* __restrict__ w,
                                                         _Float16* __restrict__ wt,
                                                         int O, int Cin){
  int i = blockIdx.x*256 + threadIdx.x;
  int tot = O*Cin*9;
  if (i >= tot) return;
  int o   = i / (Cin*9);
  int rem = i - o*(Cin*9);
  int q   = rem / Cin;
  int c   = rem - q*Cin;
  wt[i] = (_Float16)w[((size_t)o*Cin + c)*9 + q];
}

// ============================ layernorm ============================
// one block per token row; 256 threads x 4 floats = C_=1024
template<bool CLS_OUT>
__global__ __launch_bounds__(256) void ln_kernel(const float* __restrict__ x,
                                                 const float* __restrict__ g,
                                                 const float* __restrict__ bta,
                                                 _Float16* __restrict__ y16,
                                                 float* __restrict__ out){
  __shared__ float ss[8], sq[8];
  int row  = blockIdx.x;
  int t    = threadIdx.x;
  int lane = t & 31, wid = t >> 5;
  const float4 v = reinterpret_cast<const float4*>(x + (size_t)row*C_)[t];
  float s = v.x + v.y + v.z + v.w;
  float q = v.x*v.x + v.y*v.y + v.z*v.z + v.w*v.w;
  s = redsum32(s); q = redsum32(q);
  if (lane == 0){ ss[wid] = s; sq[wid] = q; }
  __syncthreads();
  float tot = 0.f, totq = 0.f;
  #pragma unroll
  for (int w = 0; w < 8; ++w){ tot += ss[w]; totq += sq[w]; }
  float mu  = tot  * (1.0f/C_);
  float var = totq * (1.0f/C_) - mu*mu;
  float rs  = rsqrtf(var + 1e-5f);
  float vv[4] = {v.x, v.y, v.z, v.w};
  bool is_cls = CLS_OUT && ((row % N_) == NSP);
  #pragma unroll
  for (int j = 0; j < 4; ++j){
    int c = t*4 + j;
    float val = (vv[j] - mu)*rs*g[c] + bta[c];
    y16[(size_t)row*C_ + c] = (_Float16)val;
    if (CLS_OUT){
      if (is_cls) out[(size_t)row*C_ + c] = vv[j] + val;   // d_out CLS row: x + LN2(x)
    }
  }
}

// ============================ generic WMMA GEMM ============================
// out[M][Nout] = A[M][K] * W[Nout][K]^T + bias (+ resid) ; block = 64M x 256N, wave = 32M x 64N
template<bool OUT_F16, bool RESID, bool GELU_>
__global__ __launch_bounds__(256) void gemm_kernel(const _Float16* __restrict__ A,
                                                   const _Float16* __restrict__ W,
                                                   const float* __restrict__ bias,
                                                   const float* __restrict__ resid,
                                                   void* __restrict__ outp,
                                                   int M, int Nout, int K){
  int lane = threadIdx.x & 31, wid = threadIdx.x >> 5;
  int wm = wid & 1, wn = wid >> 1;
  int m0 = blockIdx.x * 64  + wm * 32;
  int n0 = blockIdx.y * 256 + wn * 64;
  int lr = lane & 15, hf = lane >> 4;

  // hoisted row base pointers
  const _Float16* arow[2];
  #pragma unroll
  for (int mt = 0; mt < 2; ++mt){
    int row = m0 + mt*16 + lr; if (row > M-1) row = M-1;
    arow[mt] = A + (size_t)row*K;
  }
  const _Float16* wrow[4];
  #pragma unroll
  for (int nt = 0; nt < 4; ++nt)
    wrow[nt] = W + (size_t)(n0 + nt*16 + lr)*K;

  FA acc[2][4];
  #pragma unroll
  for (int a = 0; a < 2; ++a)
    #pragma unroll
    for (int b = 0; b < 4; ++b)
      #pragma unroll
      for (int r = 0; r < 8; ++r) acc[a][b].f[r] = 0.f;

  for (int k0 = 0; k0 < K; k0 += 32){
    AF af[2];
    #pragma unroll
    for (int mt = 0; mt < 2; ++mt){
      load_frag(af[mt], arow[mt] + k0, hf);
      if (k0 + 32 < K) __builtin_prefetch((const void*)(arow[mt] + k0 + 32), 0, 1);
    }
    #pragma unroll
    for (int nt = 0; nt < 4; ++nt){
      AF bf;
      load_frag(bf, wrow[nt] + k0, hf);
      if (k0 + 32 < K) __builtin_prefetch((const void*)(wrow[nt] + k0 + 32), 0, 1);
      #pragma unroll
      for (int mt = 0; mt < 2; ++mt)
        acc[mt][nt].v = wmma32(af[mt].h, bf.h, acc[mt][nt].v);
    }
  }

  #pragma unroll
  for (int nt = 0; nt < 4; ++nt){
    int col = n0 + nt*16 + lr;
    float bs = bias ? bias[col] : 0.f;
    #pragma unroll
    for (int mt = 0; mt < 2; ++mt){
      #pragma unroll
      for (int r = 0; r < 8; ++r){
        int row = m0 + mt*16 + r + 8*hf;
        if (row < M){
          float val = acc[mt][nt].f[r] + bs;
          if (RESID) val += resid[(size_t)row*Nout + col];
          if (GELU_) val = gelu_tanh(val);
          if (OUT_F16) ((_Float16*)outp)[(size_t)row*Nout + col] = (_Float16)val;
          else         ((float*)outp)   [(size_t)row*Nout + col] = val;
        }
      }
    }
  }
}

// ============================ flash attention ============================
// qkv: [B*N][3C] f16 (q|k|v each [H][D] inside C). One wave = 16 query rows of one (b,h).
__global__ __launch_bounds__(128) void attn_kernel(const _Float16* __restrict__ qkv,
                                                   _Float16* __restrict__ o16){
  __shared__ __align__(16) _Float16 ldsP[4][16*32];
  int lane = threadIdx.x & 31, wid = threadIdx.x >> 5;
  int bh = blockIdx.x;
  int b  = bh >> 4, h = bh & 15;
  int qt = blockIdx.y * 4 + wid;
  if (qt * 16 >= N_) return;                       // 37 q-tiles total
  int lr = lane & 15, hf = lane >> 4;
  size_t base = (size_t)b * N_;

  // Q fragments (16 rows x 64 d -> two K=32 chunks)
  AF qa[2];
  {
    int qrow = qt*16 + lr; if (qrow > N_-1) qrow = N_-1;
    const _Float16* qp = qkv + (base + qrow)*(3*C_) + h*D_;
    load_frag(qa[0], qp,      hf);
    load_frag(qa[1], qp + 32, hf);
  }

  FA O[4];
  #pragma unroll
  for (int t = 0; t < 4; ++t)
    #pragma unroll
    for (int r = 0; r < 8; ++r) O[t].f[r] = 0.f;
  float mrow[8], lsum[8];
  #pragma unroll
  for (int r = 0; r < 8; ++r){ mrow[r] = -3.0e38f; lsum[r] = 0.f; }

  for (int j0 = 0; j0 < N_; j0 += 32){
    // ---- scores: S(16x32) = Q(16x64) . K^T(64x32) * 1/8 ----
    FA S[2];
    #pragma unroll
    for (int t = 0; t < 2; ++t){
      int kc  = j0 + t*16 + lr;
      int kcc = (kc > N_-1) ? N_-1 : kc;
      const _Float16* kp = qkv + (base + kcc)*(3*C_) + C_ + h*D_;
      FA st;
      #pragma unroll
      for (int r = 0; r < 8; ++r) st.f[r] = 0.f;
      AF bf0, bf1;
      load_frag(bf0, kp,      hf);
      load_frag(bf1, kp + 32, hf);
      st.v = wmma32(qa[0].h, bf0.h, st.v);
      st.v = wmma32(qa[1].h, bf1.h, st.v);
      bool oob = (kc > N_-1);
      #pragma unroll
      for (int r = 0; r < 8; ++r) S[t].f[r] = oob ? -3.0e38f : st.f[r]*0.125f;
    }

    // ---- online softmax (row stats across the 16 lanes of each half) ----
    float alpha[8];
    #pragma unroll
    for (int r = 0; r < 8; ++r){
      float mx = redmax16(fmaxf(S[0].f[r], S[1].f[r]));
      float mn = fmaxf(mrow[r], mx);
      alpha[r] = __expf(mrow[r] - mn);
      mrow[r]  = mn;
      float p0 = __expf(S[0].f[r] - mn);
      float p1 = __expf(S[1].f[r] - mn);
      S[0].f[r] = p0; S[1].f[r] = p1;
      lsum[r] = lsum[r]*alpha[r] + redsum16(p0 + p1);
    }
    #pragma unroll
    for (int t = 0; t < 4; ++t)
      #pragma unroll
      for (int r = 0; r < 8; ++r) O[t].f[r] *= alpha[r];

    // ---- repack P: C-layout -> A-fragment layout via per-wave LDS tile ----
    _Float16* lp = ldsP[wid];
    #pragma unroll
    for (int r = 0; r < 8; ++r){
      int m = r + 8*hf;
      lp[m*32 + lr]      = (_Float16)S[0].f[r];
      lp[m*32 + 16 + lr] = (_Float16)S[1].f[r];
    }
    asm volatile("s_wait_dscnt 0" ::: "memory");   // intra-wave LDS visibility
    AF pf;
    load_frag(pf, lp + lr*32, hf);                 // ds_load_b128 x2

    // ---- V fragments: hoist the two row pointers per K-pair; dt*32B is an
    //      immediate offset on the loads (kills the per-load 64-bit addr math) ----
    AF vf[4];
    #pragma unroll
    for (int i = 0; i < 8; ++i){
      int key = j0 + koff(i, hf);
      int k0c = (key   > N_-1) ? N_-1 : key;
      int k1c = (key+1 > N_-1) ? N_-1 : key+1;
      const unsigned short* r0 = (const unsigned short*)(qkv + (base + k0c)*(3*C_) + 2*C_ + h*D_ + lr);
      const unsigned short* r1 = (const unsigned short*)(qkv + (base + k1c)*(3*C_) + 2*C_ + h*D_ + lr);
      #pragma unroll
      for (int dt = 0; dt < 4; ++dt)
        vf[dt].u[i] = (unsigned int)r0[dt*16] | ((unsigned int)r1[dt*16] << 16);
    }
    #pragma unroll
    for (int dt = 0; dt < 4; ++dt)
      O[dt].v = wmma32(pf.h, vf[dt].h, O[dt].v);
  }

  // ---- normalize and store o[b,n,h*D+d] as f16 GEMM input ----
  #pragma unroll
  for (int r = 0; r < 8; ++r){
    float inv = 1.0f / lsum[r];
    int row = qt*16 + r + 8*hf;
    if (row < N_){
      #pragma unroll
      for (int dt = 0; dt < 4; ++dt)
        o16[(base + row)*C_ + h*D_ + dt*16 + lr] = (_Float16)(O[dt].f[r]*inv);
    }
  }
}

// ============================ implicit-GEMM 3x3 conv ============================
// in: [B][tokPerB][Cin] f16 (rows indexed b*tokPerB + iy*24+ix); W: [Nout][9*Cin], k = tap*Cin + c
template<bool GELU_, bool FINAL>
__global__ __launch_bounds__(256) void conv3x3_kernel(const _Float16* __restrict__ in,
                                                      const _Float16* __restrict__ W,
                                                      const float* __restrict__ bias,
                                                      const float* __restrict__ resid,
                                                      void* __restrict__ outp,
                                                      int Cin, int tokPerB, int Nout){
  int lane = threadIdx.x & 31, wid = threadIdx.x >> 5;
  int wm = wid & 1, wn = wid >> 1;
  int m0 = blockIdx.x * 64  + wm * 32;
  int n0 = blockIdx.y * 256 + wn * 64;
  int lr = lane & 15, hf = lane >> 4;
  int K = 9 * Cin;

  int bb[2], py[2], px[2];
  #pragma unroll
  for (int mt = 0; mt < 2; ++mt){
    int m = m0 + mt*16 + lr;
    bb[mt] = m / NSP;
    int p  = m - bb[mt]*NSP;
    py[mt] = p / M_;
    px[mt] = p - py[mt]*M_;
  }
  const _Float16* wrow[4];
  #pragma unroll
  for (int nt = 0; nt < 4; ++nt)
    wrow[nt] = W + (size_t)(n0 + nt*16 + lr)*K;

  FA acc[2][4];
  #pragma unroll
  for (int a = 0; a < 2; ++a)
    #pragma unroll
    for (int b = 0; b < 4; ++b)
      #pragma unroll
      for (int r = 0; r < 8; ++r) acc[a][b].f[r] = 0.f;

  for (int k0 = 0; k0 < K; k0 += 32){
    int q  = k0 / Cin;            // Cin % 32 == 0 -> tap constant inside a K=32 chunk
    int c0 = k0 - q*Cin;
    int dy = q/3 - 1, dx = q - (q/3)*3 - 1;
    AF af[2];
    #pragma unroll
    for (int mt = 0; mt < 2; ++mt){
      int iy = py[mt] + dy, ix = px[mt] + dx;
      bool valid = (iy >= 0) && (iy < M_) && (ix >= 0) && (ix < M_);
      if (valid){
        load_frag(af[mt], in + ((size_t)(bb[mt]*tokPerB + iy*M_ + ix))*Cin + c0, hf);
      } else {
        #pragma unroll
        for (int i = 0; i < 8; ++i) af[mt].u[i] = 0u;
      }
    }
    #pragma unroll
    for (int nt = 0; nt < 4; ++nt){
      AF bf;
      load_frag(bf, wrow[nt] + k0, hf);
      if (k0 + 32 < K) __builtin_prefetch((const void*)(wrow[nt] + k0 + 32), 0, 1);
      #pragma unroll
      for (int mt = 0; mt < 2; ++mt)
        acc[mt][nt].v = wmma32(af[mt].h, bf.h, acc[mt][nt].v);
    }
  }

  #pragma unroll
  for (int nt = 0; nt < 4; ++nt){
    int col = n0 + nt*16 + lr;
    float bs = bias[col];
    #pragma unroll
    for (int mt = 0; mt < 2; ++mt){
      #pragma unroll
      for (int r = 0; r < 8; ++r){
        int m  = m0 + mt*16 + r + 8*hf;      // M = 9216 exactly, no guard needed
        float val = acc[mt][nt].f[r] + bs;
        if (GELU_) val = gelu_tanh(val);
        if (FINAL){
          int b2 = m / NSP; int p2 = m - b2*NSP;
          size_t ro = ((size_t)(b2*N_ + p2))*C_ + col;      // d_out row b*577+p
          ((float*)outp)[ro] = val + resid[ro];             // x + conv2(...)
        } else {
          ((_Float16*)outp)[(size_t)m*Nout + col] = (_Float16)val;
        }
      }
    }
  }
}

// ============================ launch ============================
extern "C" void kernel_launch(void* const* d_in, const int* in_sizes, int n_in,
                              void* d_out, int out_size, void* d_ws, size_t ws_size,
                              hipStream_t stream){
  const float* x       = (const float*)d_in[0];
  const float* ln1_g   = (const float*)d_in[1];
  const float* ln1_b   = (const float*)d_in[2];
  const float* in_w    = (const float*)d_in[3];
  const float* in_b    = (const float*)d_in[4];
  const float* out_w   = (const float*)d_in[5];
  const float* out_b   = (const float*)d_in[6];
  const float* ln2_g   = (const float*)d_in[7];
  const float* ln2_b   = (const float*)d_in[8];
  const float* conv1_w = (const float*)d_in[9];
  const float* conv1_b = (const float*)d_in[10];
  const float* conv2_w = (const float*)d_in[11];
  const float* conv2_b = (const float*)d_in[12];

  char* wsp = (char*)d_ws;
  auto alloc = [&](size_t bytes)->void*{
    void* p = (void*)wsp;
    wsp += (bytes + 255) & ~(size_t)255;
    return p;
  };
  _Float16* h16    = (_Float16*)alloc((size_t)BN*C_*2);       // LN1(x) f16
  _Float16* qkv16  = (_Float16*)alloc((size_t)BN*3*C_*2);     // qkv f16
  _Float16* o16    = (_Float16*)alloc((size_t)BN*C_*2);       // attention out f16
  float*    xbuf   = (float*)   alloc((size_t)BN*C_*4);       // x after residual f32
  _Float16* y16    = (_Float16*)alloc((size_t)BN*C_*2);       // LN2(x) f16
  _Float16* f1     = (_Float16*)alloc((size_t)MSP*CH_*2);     // conv1+gelu f16
  _Float16* inw16  = (_Float16*)alloc((size_t)3*C_*C_*2);
  _Float16* outw16 = (_Float16*)alloc((size_t)C_*C_*2);
  _Float16* wt1    = (_Float16*)alloc((size_t)CH_*9*C_*2);
  _Float16* wt2    = (_Float16*)alloc((size_t)C_*9*CH_*2);

  { int n = 3*C_*C_;  cast_f16_kernel<<<(n+255)/256,256,0,stream>>>(in_w,  inw16,  n); }
  { int n = C_*C_;    cast_f16_kernel<<<(n+255)/256,256,0,stream>>>(out_w, outw16, n); }
  { int n = CH_*C_*9; prep_convw_kernel<<<(n+255)/256,256,0,stream>>>(conv1_w, wt1, CH_, C_); }
  { int n = C_*CH_*9; prep_convw_kernel<<<(n+255)/256,256,0,stream>>>(conv2_w, wt2, C_, CH_); }

  // LN1 -> h16
  ln_kernel<false><<<BN,256,0,stream>>>(x, ln1_g, ln1_b, h16, nullptr);
  // qkv = h @ in_w^T + in_b
  gemm_kernel<true,false,false><<<dim3((BN+63)/64, (3*C_)/256),256,0,stream>>>(
      h16, inw16, in_b, nullptr, qkv16, BN, 3*C_, C_);
  // flash attention -> o16
  attn_kernel<<<dim3(B_*H_, 10),128,0,stream>>>(qkv16, o16);
  // x = o @ out_w^T + out_b + residual(x)
  gemm_kernel<false,true,false><<<dim3((BN+63)/64, C_/256),256,0,stream>>>(
      o16, outw16, out_b, x, xbuf, BN, C_, C_);
  // LN2 -> y16, plus d_out CLS rows = x + LN2(x)
  ln_kernel<true><<<BN,256,0,stream>>>(xbuf, ln2_g, ln2_b, y16, (float*)d_out);
  // conv1 + gelu -> f1
  conv3x3_kernel<true,false><<<dim3(MSP/64, CH_/256),256,0,stream>>>(
      y16, wt1, conv1_b, nullptr, f1, C_, N_, CH_);
  // conv2 + bias + residual(x) -> d_out spatial rows
  conv3x3_kernel<false,true><<<dim3(MSP/64, C_/256),256,0,stream>>>(
      f1, wt2, conv2_b, xbuf, d_out, CH_, NSP, C_);
}
